// NeuralNetwork_64690797413052
// MI455X (gfx1250) — matrix operand order repaired and checked by reading, not compiled, not run on hardware
//
#include <hip/hip_runtime.h>

// ---------------------------------------------------------------------------
// DEQ + Anderson acceleration for MI455X (gfx1250, wave32).
//   B=16384, S=31, D=14, H=50, dL=434, 6-slot history, 48 solver iterations.
// Roofline: ~100 GFLOP compute vs ~19 GB of sequential history traffic
// -> HBM-bound (~0.85 ms @ 23.3 TB/s). Strategy:
//   * precompute hx = x@Wh^T + bh + bx once (101.6 MB, fits 192 MB L2) and
//     keep it hot; stream G/F history with non-temporal loads/stores,
//   * f(z) GEMMs via v_wmma_f32_16x16x32_bf16 (fp32 accumulate),
//   * deterministic 2-stage norm reduction (graph-replay stable).
// ---------------------------------------------------------------------------

#define NB     16384          // batch
#define SEQ    31             // S
#define DD     14             // D
#define HH     50             // H
#define DL     434            // S*D
#define SLOT   2604           // 6*DL  (per-batch history stride)
#define NROW   507904         // NB*SEQ rows of the flattened GEMMs
#define NTILE  31744          // NROW/16
#define NBE    7110656        // NB*DL
#define FBLK   3968           // fstep blocks (8 waves/block, 1 tile/wave)
#define LAMREG 1e-4f

typedef __attribute__((ext_vector_type(16))) __bf16 v16bf;
typedef __attribute__((ext_vector_type(8)))  float  v8f;

union BfPack { unsigned int u[8]; v16bf v; };

static __device__ __forceinline__ unsigned int f2bf(float f) {
  unsigned int u = __float_as_uint(f);
  u += 0x7FFFu + ((u >> 16) & 1u);          // round-to-nearest-even
  return u >> 16;
}
static __device__ __forceinline__ unsigned int packbf(float a, float b) {
  return (f2bf(a) & 0xFFFFu) | (f2bf(b) << 16);
}

// ---------------------------------------------------------------------------
// hx = x @ Wh^T + (bh + bx)   [NROW,50] fp32 (one-time; stays L2-resident)
// ---------------------------------------------------------------------------
__global__ __launch_bounds__(256) void hx_kernel(
    const float* __restrict__ x, const float* __restrict__ Wh,
    const float* __restrict__ bh, const float* __restrict__ bx,
    float* __restrict__ hx) {
  size_t t = (size_t)blockIdx.x * 256 + threadIdx.x;   // exact grid
  int r = (int)(t / HH);
  int h = (int)(t - (size_t)r * HH);
  const float* xr = x + (size_t)r * DD;
  const float* wr = Wh + h * DD;
  float s = bh[h] + bx[h];
#pragma unroll
  for (int d = 0; d < DD; ++d) s = fmaf(xr[d], wr[d], s);
  hx[t] = s;
}

// ---------------------------------------------------------------------------
// fstep: F = tanh( tanh(Z@Wx^T + hx) @ Wo^T + bo ), optional G = F - Z and
// deterministic per-block partials of ||G||^2, ||F||^2.
// One wave computes one 16-row tile with 6 v_wmma_f32_16x16x32_bf16.
// ---------------------------------------------------------------------------
__global__ __launch_bounds__(256) void fstep_kernel(
    const float* __restrict__ zsrc, int zBS, int zOff,
    float* __restrict__ fdst, int fBS, int fOff,
    float* __restrict__ gdst,
    const float* __restrict__ hx,
    const float* __restrict__ Wx,
    const float* __restrict__ Wo,
    const float* __restrict__ bo,
    float* __restrict__ partial) {
  __shared__ __align__(16) unsigned short ldsH[8][16][68]; // per-wave hidden tile (bf16)
  __shared__ float red[16];

  const int wave = threadIdx.x >> 5;
  const int lane = threadIdx.x & 31;
  const int hl   = lane >> 4;        // lane half (A/B operand K split)
  const int l16  = lane & 15;
  const int tile = blockIdx.x * 8 + wave;   // grid is exact: tile < NTILE

  float d2acc = 0.f, f2acc = 0.f;

  // ---- A operand: 16x32 bf16 tile of Z (K = D = 14, zero padded to 32).
  // ISA A layout: lanes 0-15 hold K 0-7 & 16-23, lanes 16-31 hold K 8-15 & 24-31.
  const int rowA = tile * 16 + l16;
  const int bA = rowA / SEQ, sA = rowA - bA * SEQ;
  const float* zrow = zsrc + (size_t)bA * zBS + zOff + sA * DD;
  BfPack A1;
#pragma unroll
  for (int v = 0; v < 8; ++v) {
    if (v < 4) {
      int k0 = hl * 8 + 2 * v;
      float a0 = (k0     < DD) ? zrow[k0]     : 0.f;
      float a1 = (k0 + 1 < DD) ? zrow[k0 + 1] : 0.f;
      A1.u[v] = packbf(a0, a1);
    } else {
      A1.u[v] = 0u;                 // K >= 16 is zero padding
    }
  }

  // ---- GEMM1: hidden = tanh(Z@Wx^T + hx); N = H = 50 padded to 4x16.
#pragma unroll
  for (int t = 0; t < 4; ++t) {
    const int colN = t * 16 + l16;
    BfPack B1;                       // B layout: lanes 0-15 K 0-7 & 16-23, lanes 16-31 K 8-15 & 24-31
#pragma unroll
    for (int v = 0; v < 8; ++v) {
      unsigned int u = 0u;
      if (v < 4 && colN < HH) {
        int k0 = hl * 8 + 2 * v;
        float b0 = (k0     < DD) ? Wx[colN * DD + k0]     : 0.f;
        float b1 = (k0 + 1 < DD) ? Wx[colN * DD + k0 + 1] : 0.f;
        u = packbf(b0, b1);
      }
      B1.u[v] = u;
    }
    v8f c;                           // C init = hx  (bias adds ride the WMMA)
#pragma unroll
    for (int r = 0; r < 8; ++r) {
      int row = tile * 16 + r + hl * 8;
      c[r] = (colN < HH) ? hx[(size_t)row * HH + colN] : 0.f;
    }
    c = __builtin_amdgcn_wmma_f32_16x16x32_bf16(false, A1.v, false, B1.v,
                                                (short)0, c, false, false);
#pragma unroll
    for (int r = 0; r < 8; ++r)      // transpose C-layout -> A-layout via LDS
      ldsH[wave][r + hl * 8][colN] = (unsigned short)f2bf(tanhf(c[r]));
  }

  // ---- GEMM2: out = tanh(hidden @ Wo^T + bo); K = 50 padded to 64, N = 14.
  const int colO = l16;
  v8f c2;
#pragma unroll
  for (int r = 0; r < 8; ++r) c2[r] = (colO < DD) ? bo[colO] : 0.f;
#pragma unroll
  for (int ch = 0; ch < 2; ++ch) {
    BfPack A2, B2;
#pragma unroll
    for (int v = 0; v < 8; ++v) {
      int koff = ch * 32 + (v >> 2) * 16 + hl * 8 + (v & 3) * 2;
      A2.u[v] = *(const unsigned int*)&ldsH[wave][l16][koff];
      int kb = ch * 32 + (v >> 2) * 16 + hl * 8 + (v & 3) * 2;
      float b0 = (colO < DD && kb     < HH) ? Wo[colO * HH + kb]     : 0.f;
      float b1 = (colO < DD && kb + 1 < HH) ? Wo[colO * HH + kb + 1] : 0.f;
      B2.u[v] = packbf(b0, b1);
    }
    c2 = __builtin_amdgcn_wmma_f32_16x16x32_bf16(false, A2.v, false, B2.v,
                                                 (short)0, c2, false, false);
  }

  // ---- store F (+G), accumulate norm partials (non-temporal: don't evict hx)
#pragma unroll
  for (int r = 0; r < 8; ++r) {
    int row = tile * 16 + r + hl * 8;
    int bb = row / SEQ, ss = row - bb * SEQ;
    float fv = tanhf(c2[r]);
    if (colO < DD) {
      size_t o = (size_t)bb * fBS + fOff + ss * DD + colO;
      __builtin_nontemporal_store(fv, &fdst[o]);
      if (gdst != nullptr || partial != nullptr) {
        float xv = zsrc[(size_t)bb * zBS + zOff + ss * DD + colO];
        float gv = fv - xv;
        if (gdst != nullptr) __builtin_nontemporal_store(gv, &gdst[o]);
        d2acc += gv * gv;
        f2acc += fv * fv;
      }
    }
  }

  if (partial != nullptr) {          // deterministic block reduction
#pragma unroll
    for (int off = 16; off > 0; off >>= 1) {
      d2acc += __shfl_xor(d2acc, off);
      f2acc += __shfl_xor(f2acc, off);
    }
    if (lane == 0) { red[wave] = d2acc; red[8 + wave] = f2acc; }
    __syncthreads();
    if (threadIdx.x == 0) {
      float a = 0.f, b = 0.f;
#pragma unroll
      for (int w = 0; w < 8; ++w) { a += red[w]; b += red[8 + w]; }
      partial[blockIdx.x] = a;
      partial[gridDim.x + blockIdx.x] = b;
    }
  }
}

// ---------------------------------------------------------------------------
// Anderson step: GG = G G^T (+lam I), solve GG y = 1 (Cholesky), alpha = y/Sum,
// Xcur = sum_i alpha_i F_i.   One wave per batch element.
// (Equivalent to the reference bordered (n+1)x(n+1) solve with BETA = 1.)
// ---------------------------------------------------------------------------
__global__ __launch_bounds__(256) void anderson_kernel(
    const float* __restrict__ G, const float* __restrict__ F,
    float* __restrict__ Xcur, int n) {
  const int wave = threadIdx.x >> 5;
  const int lane = threadIdx.x & 31;
  const int b = blockIdx.x * 8 + wave;     // exact grid
  const size_t base = (size_t)b * SLOT;

  float s[6][6];
#pragma unroll
  for (int i = 0; i < 6; ++i)
#pragma unroll
    for (int j = 0; j < 6; ++j) s[i][j] = 0.f;

  for (int c = 0; c < 14; ++c) {
    int e = lane + 32 * c;
    bool act = e < DL;
    float g[6];
#pragma unroll
    for (int i = 0; i < 6; ++i)
      g[i] = (i < n && act) ? __builtin_nontemporal_load(&G[base + i * DL + e]) : 0.f;
#pragma unroll
    for (int i = 0; i < 6; ++i)
#pragma unroll
      for (int j = i; j < 6; ++j)
        if (j < n) s[i][j] = fmaf(g[i], g[j], s[i][j]);
  }
#pragma unroll
  for (int i = 0; i < 6; ++i)
#pragma unroll
    for (int j = i; j < 6; ++j)
      if (j < n)
#pragma unroll
        for (int off = 16; off > 0; off >>= 1) s[i][j] += __shfl_xor(s[i][j], off);

  float gg[6][6];
#pragma unroll
  for (int i = 0; i < 6; ++i)
#pragma unroll
    for (int j = 0; j < 6; ++j)
      gg[i][j] = ((i <= j) ? s[i][j] : s[j][i]) + ((i == j) ? LAMREG : 0.f);

  float L[6][6];
#pragma unroll
  for (int i = 0; i < 6; ++i)
#pragma unroll
    for (int j = 0; j < 6; ++j) L[i][j] = 0.f;
#pragma unroll
  for (int i = 0; i < 6; ++i) {
    if (i < n) {
      float d = gg[i][i];
#pragma unroll
      for (int p = 0; p < 6; ++p) if (p < i) d -= L[i][p] * L[i][p];
      d = sqrtf(fmaxf(d, 1e-30f));
      L[i][i] = d;
      float inv = 1.f / d;
#pragma unroll
      for (int r = 0; r < 6; ++r) {
        if (r > i && r < n) {
          float t = gg[r][i];
#pragma unroll
          for (int p = 0; p < 6; ++p) if (p < i) t -= L[r][p] * L[i][p];
          L[r][i] = t * inv;
        }
      }
    }
  }
  float wv[6];
#pragma unroll
  for (int i = 0; i < 6; ++i) {     // forward: L w = 1
    if (i < n) {
      float t = 1.f;
#pragma unroll
      for (int p = 0; p < 6; ++p) if (p < i) t -= L[i][p] * wv[p];
      wv[i] = t / L[i][i];
    } else wv[i] = 0.f;
  }
  float y[6];
#pragma unroll
  for (int i = 5; i >= 0; --i) {    // backward: L^T y = w
    if (i < n) {
      float t = wv[i];
#pragma unroll
      for (int p = 0; p < 6; ++p) if (p > i && p < n) t -= L[p][i] * y[p];
      y[i] = t / L[i][i];
    } else y[i] = 0.f;
  }
  float sum = 0.f;
#pragma unroll
  for (int i = 0; i < 6; ++i) sum += y[i];
  float inv = 1.f / sum;
  float alpha[6];
#pragma unroll
  for (int i = 0; i < 6; ++i) alpha[i] = y[i] * inv;

  for (int c = 0; c < 14; ++c) {
    int e = lane + 32 * c;
    if (e < DL) {
      float acc = 0.f;
#pragma unroll
      for (int i = 0; i < 6; ++i)
        if (i < n) acc = fmaf(alpha[i], __builtin_nontemporal_load(&F[base + i * DL + e]), acc);
      __builtin_nontemporal_store(acc, &Xcur[(size_t)b * DL + e]);
    }
  }
}

// ---------------------------------------------------------------------------
__global__ __launch_bounds__(256) void init_kernel(float* Xc, float* LX,
                                                   float* lowest, int* flag) {
  size_t i = (size_t)blockIdx.x * 256 + threadIdx.x;   // exact grid (NBE)
  Xc[i] = 0.f;
  LX[i] = 0.f;
  if (i == 0) { *lowest = 1e8f; *flag = 0; }
}

__global__ __launch_bounds__(256) void seed_kernel(const float* __restrict__ F,
                                                   float* __restrict__ Xc) {
  size_t i = (size_t)blockIdx.x * 256 + threadIdx.x;   // Xcur = F slot 0
  int b = (int)(i / DL);
  int e = (int)(i - (size_t)b * DL);
  Xc[i] = F[(size_t)b * SLOT + e];
}

__global__ __launch_bounds__(256) void scalar_kernel(const float* __restrict__ partial,
                                                     float* lowest, int* flag) {
  __shared__ float rd[512];
  float a = 0.f, b = 0.f;
  for (int i = threadIdx.x; i < FBLK; i += 256) { a += partial[i]; b += partial[FBLK + i]; }
  rd[threadIdx.x] = a;
  rd[256 + threadIdx.x] = b;
  __syncthreads();
  for (int st = 128; st > 0; st >>= 1) {
    if ((int)threadIdx.x < st) {
      rd[threadIdx.x] += rd[threadIdx.x + st];
      rd[256 + threadIdx.x] += rd[256 + threadIdx.x + st];
    }
    __syncthreads();
  }
  if (threadIdx.x == 0) {
    float rel = sqrtf(rd[0]) / (1e-5f + sqrtf(rd[256]));
    int better = (rel < *lowest) ? 1 : 0;
    *flag = better;
    if (better) *lowest = rel;
  }
}

__global__ __launch_bounds__(256) void select_kernel(const float* __restrict__ Xc,
                                                     float* __restrict__ LX,
                                                     const int* __restrict__ flag) {
  if (*flag == 0) return;
  size_t i = (size_t)blockIdx.x * 256 + threadIdx.x;
  LX[i] = Xc[i];
}

__global__ __launch_bounds__(256) void out_kernel(const float* __restrict__ z,
                                                  const float* __restrict__ Wf,
                                                  const float* __restrict__ bf,
                                                  float* __restrict__ out) {
  int wave = threadIdx.x >> 5, lane = threadIdx.x & 31;
  int b = blockIdx.x * 8 + wave;     // exact grid
  float s = 0.f;
  for (int c = 0; c < 14; ++c) {
    int e = lane + 32 * c;
    if (e < DL) s = fmaf(z[(size_t)b * SLOT + e], Wf[e], s);
  }
#pragma unroll
  for (int off = 16; off > 0; off >>= 1) s += __shfl_xor(s, off);
  if (lane == 0) out[b] = s + bf[0];
}

// ---------------------------------------------------------------------------
extern "C" void kernel_launch(void* const* d_in, const int* in_sizes, int n_in,
                              void* d_out, int out_size, void* d_ws, size_t ws_size,
                              hipStream_t stream) {
  const float* x  = (const float*)d_in[0];
  const float* Wh = (const float*)d_in[1];
  const float* bh = (const float*)d_in[2];
  const float* Wx = (const float*)d_in[3];
  const float* bx = (const float*)d_in[4];
  const float* Wo = (const float*)d_in[5];
  const float* bo = (const float*)d_in[6];
  const float* Wf = (const float*)d_in[7];
  const float* bf = (const float*)d_in[8];
  float* out = (float*)d_out;

  float* ws = (float*)d_ws;
  float* hx   = ws;                                  // [NROW*50]      101.6 MB
  float* G    = hx + (size_t)NROW * HH;              // [NB*6*DL]      170.7 MB
  float* F    = G + (size_t)NB * SLOT;               // [NB*6*DL]      170.7 MB
  float* Xc   = F + (size_t)NB * SLOT;               // [NB*DL]         28.4 MB
  float* LX   = Xc + (size_t)NBE;                    // [NB*DL]         28.4 MB
  float* PART = LX + (size_t)NBE;                    // [2*FBLK]
  float* lowest = PART + 2 * FBLK;
  int*   flag   = (int*)(lowest + 1);

  // precompute hx (L2-resident across all iterations)
  hx_kernel<<<99200, 256, 0, stream>>>(x, Wh, bh, bx, hx);
  // Xcur = 0, lowest_x = 0, lowest = 1e8, flag = 0
  init_kernel<<<27776, 256, 0, stream>>>(Xc, LX, lowest, flag);

  // k=0: F0 = f(0), G0 = F0 - 0
  fstep_kernel<<<FBLK, 256, 0, stream>>>(Xc, DL, 0, F, SLOT, 0, G,
                                         hx, Wx, Wo, bo, nullptr);
  // k=1: X1 = F0; F1 = f(X1), G1 = F1 - X1
  seed_kernel<<<27776, 256, 0, stream>>>(F, Xc);
  fstep_kernel<<<FBLK, 256, 0, stream>>>(Xc, DL, 0, F, SLOT, DL, G,
                                         hx, Wx, Wo, bo, nullptr);

  for (int k = 2; k < 50; ++k) {
    int n = (k < 6) ? k : 6;
    int j = k % 6;
    anderson_kernel<<<2048, 256, 0, stream>>>(G, F, Xc, n);
    fstep_kernel<<<FBLK, 256, 0, stream>>>(Xc, DL, 0, F, SLOT, j * DL, G,
                                           hx, Wx, Wo, bo, PART);
    scalar_kernel<<<1, 256, 0, stream>>>(PART, lowest, flag);
    select_kernel<<<27776, 256, 0, stream>>>(Xc, LX, flag);
  }

  // z = f(lowest_x)  (written into the now-free G slot-0 region)
  fstep_kernel<<<FBLK, 256, 0, stream>>>(LX, DL, 0, G, SLOT, 0, nullptr,
                                         hx, Wx, Wo, bo, nullptr);
  // out = z @ Wf^T + bf
  out_kernel<<<2048, 256, 0, stream>>>(G, Wf, bf, out);
}
